// PointNetSetAbstraction_4011499454796
// MI455X (gfx1250) — matrix operand order repaired and checked by reading, not compiled, not run on hardware
//
#include <hip/hip_runtime.h>
#include <hip/hip_bf16.h>
#include <stdint.h>

typedef __attribute__((ext_vector_type(16))) _Float16 v16h;
typedef __attribute__((ext_vector_type(8)))  _Float16 v8h;
typedef __attribute__((ext_vector_type(8)))  float    v8f;

#define NB 16
#define NPTS 4096
#define CIN 128
#define NPOINT 1024
#define NSAMPLE 32
#define MTOT (NPOINT*NSAMPLE)   /* 32768 */
#define NMT (MTOT/16)           /* 2048 m-tiles */
#define BN_EPS 1e-5f

// Tiled activation layout ("WMMA fragment layout"):
//   buffer[b][ks][mt] = 1KB tile, ks = ci/32, mt = m/16
//   within tile: chunk of 16 f16 per lane, lane = 16*((ci>>4)&1) + (m&15), element = ci&15
// A single lane's B-fragment (v16h) is one contiguous 32B chunk; a producer lane's
// half-D-fragment (v8h) is one contiguous 16B chunk. Loads/stores vectorize fully.

// ---------------- 1) Furthest point sampling: one block per batch ----------------
__global__ void fps_kernel(const float* __restrict__ xyz, float* __restrict__ out_newxyz)
{
  int b = blockIdx.x;
  int tid = threadIdx.x; // 256 threads, each owns 16 points in registers
  const float* base = xyz + (size_t)b * NPTS * 3;

  float px[16], py[16], pz[16], mind[16];
  #pragma unroll
  for (int j = 0; j < 16; ++j) {
    int n = j * 256 + tid;
    px[j] = base[n*3+0]; py[j] = base[n*3+1]; pz[j] = base[n*3+2];
    mind[j] = 1e10f;
  }

  __shared__ float cent[3];
  __shared__ float rv[256];
  __shared__ int   ri[256];
  __shared__ int   sfar;
  if (tid == 0) sfar = 0;
  __syncthreads();

  for (int s = 0; s < NPOINT; ++s) {
    int far = sfar;
    if (tid == 0) {
      float cx = base[far*3+0], cy = base[far*3+1], cz = base[far*3+2];
      cent[0]=cx; cent[1]=cy; cent[2]=cz;
      float* o = out_newxyz + ((size_t)b*NPOINT + s)*3;
      o[0]=cx; o[1]=cy; o[2]=cz;
    }
    __syncthreads();
    float cx=cent[0], cy=cent[1], cz=cent[2];
    float bv = -1.0f; int bi = 0x7fffffff;
    #pragma unroll
    for (int j = 0; j < 16; ++j) {
      float dx = px[j]-cx, dy = py[j]-cy, dz = pz[j]-cz;
      float d = dx*dx + dy*dy + dz*dz;
      float md = mind[j] < d ? mind[j] : d;
      mind[j] = md;
      int n = j*256 + tid;
      if (md > bv || (md == bv && n < bi)) { bv = md; bi = n; }
    }
    rv[tid]=bv; ri[tid]=bi;
    __syncthreads();
    for (int st = 128; st > 0; st >>= 1) {
      if (tid < st) {
        if (rv[tid+st] > rv[tid] || (rv[tid+st]==rv[tid] && ri[tid+st] < ri[tid])) {
          rv[tid]=rv[tid+st]; ri[tid]=ri[tid+st];
        }
      }
      __syncthreads();
    }
    if (tid == 0) sfar = ri[0];
    __syncthreads();
  }
}

// ---------------- 2) KNN: one wave (32 threads) per centroid ----------------
__global__ void knn_kernel(const float* __restrict__ xyz, const float* __restrict__ newxyz,
                           int* __restrict__ idxout)
{
  __shared__ float dist[NPTS];   // 16KB
  int lane = threadIdx.x;        // 32 threads
  int gs = blockIdx.x;           // b*NPOINT + s
  int b = gs >> 10;
  const float* base = xyz + (size_t)b * NPTS * 3;
  const float* c = newxyz + (size_t)gs * 3;
  float cx=c[0], cy=c[1], cz=c[2];
  for (int j = 0; j < NPTS/32; ++j) {
    int n = j*32 + lane;
    float dx = base[n*3+0]-cx, dy = base[n*3+1]-cy, dz = base[n*3+2]-cz;
    dist[n] = dx*dx+dy*dy+dz*dz;
  }
  __syncthreads();
  int* outp = idxout + (size_t)gs * NSAMPLE;
  for (int k = 0; k < NSAMPLE; ++k) {
    float bv = 3.0e38f; int bi = 0x7fffffff;
    for (int j = 0; j < NPTS/32; ++j) {
      int n = j*32 + lane;
      float d = dist[n];
      if (d < bv || (d == bv && n < bi)) { bv = d; bi = n; }
    }
    for (int off = 16; off > 0; off >>= 1) {
      float ov = __shfl_xor(bv, off, 32);
      int   oi = __shfl_xor(bi, off, 32);
      if (ov < bv || (ov == bv && oi < bi)) { bv = ov; bi = oi; }
    }
    if (lane == 0) { outp[k] = bi; dist[bi] = 3.0e38f; }
    __syncthreads();
  }
}

// ---------------- 3) Gather + transpose into tiled fragment layout ----------------
__global__ void gather_kernel(const float* __restrict__ points, const int* __restrict__ idx,
                              _Float16* __restrict__ FA)
{
  __shared__ _Float16 tile[32][130];     // 32 m x 128 ci, padded
  int tid = threadIdx.x;                 // 256
  int blk = blockIdx.x;
  int b    = blk >> 10;                  // / (MTOT/32)
  int mg32 = blk & 1023;
  int m0 = mg32 * 32;
  int ci = tid & 127, msub = tid >> 7;
  const int*   ib = idx + (size_t)b * MTOT + m0;
  const float* pb = points + (size_t)b * NPTS * CIN;
  #pragma unroll
  for (int p = 0; p < 16; ++p) {
    int ml = p*2 + msub;
    int n  = ib[ml];
    tile[ml][ci] = (_Float16)pb[(size_t)n * CIN + ci];
  }
  __syncthreads();
  int cg  = tid >> 5;                    // 0..7 : 16-channel group
  int m_l = tid & 31;
  int ks = cg >> 1, halfc = cg & 1;
  int mt_g = (m0 >> 4) + (m_l >> 4);
  int lane = halfc*16 + (m_l & 15);
  v16h v;
  #pragma unroll
  for (int i = 0; i < 16; ++i) v[i] = tile[m_l][cg*16 + i];
  *(v16h*)(FA + ((((size_t)b*4 + ks)*NMT + mt_g)*32 + lane)*16) = v;
}

// ---------------- 4) Pack W (f32 [co][ci]) into A-fragment layout (f16) ----------------
__global__ void wprep_kernel(const float* __restrict__ W, _Float16* __restrict__ Wp)
{
  int t = blockIdx.x*blockDim.x + threadIdx.x;   // Cout*128 threads total
  int co = t >> 7, k = t & 127;
  int ct = co >> 4, lr = co & 15;
  int ks = k >> 5, kin = k & 31;
  int half = (kin >> 4) & 1;
  int e = kin & 15;   // invert K = e+16*half
  int lane = half*16 + lr;
  Wp[(((size_t)ct*4 + ks)*32 + lane)*16 + e] = (_Float16)W[t];
}

// ---------------- 5) WMMA matmul + bias, tiled in / tiled out ----------------
__global__ void mm_kernel(const _Float16* __restrict__ Wp, const float* __restrict__ bias,
                          const _Float16* __restrict__ Fin, _Float16* __restrict__ Yout,
                          int Cout)
{
  const int coTiles = Cout >> 4;
  const int nMG = MTOT / 64;                 // 512 groups of 64 columns
  int lane = threadIdx.x & 31;
  int gid  = blockIdx.x * (blockDim.x >> 5) + (threadIdx.x >> 5);
  int b   = gid / (coTiles * nMG);
  int rem = gid % (coTiles * nMG);
  int ct  = rem / nMG;
  int mg  = rem % nMG;
  int half = lane >> 4;
  int lr   = lane & 15;

  v16h afrag[4];
  #pragma unroll
  for (int ks = 0; ks < 4; ++ks)
    afrag[ks] = *(const v16h*)(Wp + (((size_t)ct*4 + ks)*32 + lane)*16);

  float bvv[8];
  #pragma unroll
  for (int r = 0; r < 8; ++r) bvv[r] = bias[(ct<<4) + r + half*8];

  const _Float16* fb0 = Fin + (size_t)b * 4 * NMT * 512;      // Cin = 128 -> 4 ks blocks
  _Float16* yb0 = Yout + ((size_t)b * (Cout>>5) + (ct>>1)) * NMT * 512;
  int lane_c = ((ct & 1) << 4) + lr;

  #pragma unroll
  for (int mt = 0; mt < 4; ++mt) {
    int mt_g = (mg << 2) + mt;
    if (mt < 3)
      __builtin_prefetch(fb0 + (((size_t)(mt_g+1))*32 + lane)*16, 0, 0);
    v8f acc = {0.f,0.f,0.f,0.f,0.f,0.f,0.f,0.f};
    #pragma unroll
    for (int ks = 0; ks < 4; ++ks) {
      v16h bfrag = *(const v16h*)(fb0 + (((size_t)ks*NMT + mt_g)*32 + lane)*16);
      acc = __builtin_amdgcn_wmma_f32_16x16x32_f16(false, afrag[ks], false, bfrag,
                                                   (short)0, acc, false, false);
    }
    v8h o;
    #pragma unroll
    for (int r = 0; r < 8; ++r) o[r] = (_Float16)(acc[r] + bvv[r]);
    *(v8h*)(yb0 + (((size_t)mt_g)*32 + lane_c)*16 + half*8) = o;
  }
}

// ---------------- 6) Per-channel sum / sumsq over (B, M), tiled layout ----------------
__global__ void zero_kernel(float* __restrict__ p, int n)
{
  int i = blockIdx.x*blockDim.x + threadIdx.x;
  if (i < n) p[i] = 0.f;
}

__global__ void stats_kernel(const _Float16* __restrict__ Y, int CoutBlocks,
                             float* __restrict__ stats)
{
  // grid = CoutBlocks * 128 chunks; each 16-lane group reads contiguous 32B runs
  __shared__ float l1[2][16][16], l2[2][16][16];   // [h][sub][e]
  int tid = threadIdx.x;
  int ks    = blockIdx.x >> 7;
  int chunk = blockIdx.x & 127;
  int e = tid & 15, sub = tid >> 4;
  float a1[2] = {0.f,0.f}, a2[2] = {0.f,0.f};
  for (int t = 0; t < 256; ++t) {
    int tt = chunk*256 + t;                 // (b,mt) in [0, 16*2048)
    int b  = tt >> 11;
    int mt = tt & (NMT-1);
    const _Float16* base = Y + (((size_t)b*CoutBlocks + ks)*NMT + mt)*512;
    #pragma unroll
    for (int h = 0; h < 2; ++h) {
      float v = (float)base[(h*16+sub)*16 + e];
      a1[h] += v; a2[h] += v*v;
    }
  }
  l1[0][sub][e]=a1[0]; l1[1][sub][e]=a1[1];
  l2[0][sub][e]=a2[0]; l2[1][sub][e]=a2[1];
  __syncthreads();
  if (tid < 32) {
    int h = tid >> 4, ee = tid & 15;
    float s1=0.f, s2=0.f;
    #pragma unroll
    for (int s=0;s<16;++s){ s1+=l1[h][s][ee]; s2+=l2[h][s][ee]; }
    int c = ks*32 + h*16 + ee;
    atomicAdd(&stats[c],       s1);
    atomicAdd(&stats[256 + c], s2);
  }
}

// ---------------- 7) BN coefficient prep + in-place BN+ReLU (tiled) ----------------
__global__ void bnprep_kernel(const float* __restrict__ stats, const float* __restrict__ g,
                              const float* __restrict__ bb, float* __restrict__ coef, int Cout)
{
  int c = threadIdx.x;
  if (c < Cout) {
    const float inv = 1.0f / (float)(NB * MTOT);
    float mean = stats[c] * inv;
    float var  = stats[256+c] * inv - mean*mean;
    float sc = g[c] * rsqrtf(var + BN_EPS);
    coef[c]       = sc;
    coef[256 + c] = bb[c] - mean * sc;
  }
}

__global__ void bn_kernel(_Float16* __restrict__ Y, const float* __restrict__ coef,
                          int CoutBlocks)
{
  size_t r = (size_t)blockIdx.x * blockDim.x + threadIdx.x;  // 32B run id
  int rr = (int)(r & 31);
  size_t tile = r >> 5;
  int ks = (int)((tile >> 11) % (size_t)CoutBlocks);         // tile / NMT mod blocks
  int ci0 = ks*32 + ((rr >> 4) << 4);
  _Float16* p = Y + r*16;
  v16h x = *(const v16h*)p;
  v8f sc0 = *(const v8f*)(coef + ci0);
  v8f sc1 = *(const v8f*)(coef + ci0 + 8);
  v8f sh0 = *(const v8f*)(coef + 256 + ci0);
  v8f sh1 = *(const v8f*)(coef + 256 + ci0 + 8);
  v16h o;
  #pragma unroll
  for (int i = 0; i < 8; ++i) {
    float y0 = sc0[i]*(float)x[i]   + sh0[i];
    float y1 = sc1[i]*(float)x[i+8] + sh1[i];
    o[i]   = (_Float16)(y0 > 0.f ? y0 : 0.f);
    o[i+8] = (_Float16)(y1 > 0.f ? y1 : 0.f);
  }
  *(v16h*)p = o;
}

// ---------------- 8) Pool: per-(b,c) sums -> mu, E[f^2]-2mu^2+1e-6 ----------------
__global__ void poolacc_kernel(const _Float16* __restrict__ F, float* __restrict__ pstat)
{
  // grid = NB * 8 * 16 ; Cfinal = 256 -> 8 ks blocks
  __shared__ float l1[2][16][16], l2[2][16][16];
  int tid = threadIdx.x;
  int blk = blockIdx.x;
  int b     = blk >> 7;
  int ks    = (blk >> 4) & 7;
  int chunk = blk & 15;
  int e = tid & 15, sub = tid >> 4;
  float a1[2] = {0.f,0.f}, a2[2] = {0.f,0.f};
  const _Float16* base0 = F + ((size_t)b*8 + ks)*NMT*512;
  for (int t = 0; t < 128; ++t) {
    int mt = chunk*128 + t;
    const _Float16* base = base0 + (size_t)mt*512;
    #pragma unroll
    for (int h = 0; h < 2; ++h) {
      float v = (float)base[(h*16+sub)*16 + e];
      a1[h] += v; a2[h] += v*v;
    }
  }
  l1[0][sub][e]=a1[0]; l1[1][sub][e]=a1[1];
  l2[0][sub][e]=a2[0]; l2[1][sub][e]=a2[1];
  __syncthreads();
  if (tid < 32) {
    int h = tid >> 4, ee = tid & 15;
    float s1=0.f, s2=0.f;
    #pragma unroll
    for (int s=0;s<16;++s){ s1+=l1[h][s][ee]; s2+=l2[h][s][ee]; }
    int c = ks*32 + h*16 + ee;
    atomicAdd(&pstat[((size_t)b*256 + c)*2 + 0], s1);
    atomicAdd(&pstat[((size_t)b*256 + c)*2 + 1], s2);
  }
}

__global__ void poolfin_kernel(const float* __restrict__ pstat, float* __restrict__ out_tf)
{
  int b = blockIdx.x, c = threadIdx.x;
  float s1 = pstat[((size_t)b*256+c)*2+0];
  float s2 = pstat[((size_t)b*256+c)*2+1];
  float mu = s1 / (float)MTOT;
  float vt = s2 / (float)MTOT - 2.f*mu*mu + 1e-6f;
  out_tf[((size_t)b*256+c)*2+0] = mu;
  out_tf[((size_t)b*256+c)*2+1] = vt;
}

extern "C" void kernel_launch(void* const* d_in, const int* in_sizes, int n_in,
                              void* d_out, int out_size, void* d_ws, size_t ws_size,
                              hipStream_t stream)
{
  const float* xyz    = (const float*)d_in[0];
  const float* points = (const float*)d_in[1];
  const float* W0=(const float*)d_in[2],  *b0=(const float*)d_in[3],
             * g0=(const float*)d_in[4],  *bb0=(const float*)d_in[5];
  const float* W1=(const float*)d_in[6],  *b1=(const float*)d_in[7],
             * g1=(const float*)d_in[8],  *bb1=(const float*)d_in[9];
  const float* W2=(const float*)d_in[10], *b2=(const float*)d_in[11],
             * g2=(const float*)d_in[12], *bb2=(const float*)d_in[13];
  float* out = (float*)d_out;

  // workspace layout (bytes)
  char* ws = (char*)d_ws;
  _Float16* FA  = (_Float16*)(ws);                          // 134217728
  _Float16* FB  = (_Float16*)(ws + 134217728ull);           // 268435456
  int*      idx = (int*)     (ws + 402653184ull);           //   2097152
  float*  stats = (float*)   (ws + 404750336ull);           //      2048
  float*  coef  = (float*)   (ws + 404752384ull);           //      2048
  float*  pstat = (float*)   (ws + 404754432ull);           //     32768
  _Float16* wp0 = (_Float16*)(ws + 404787200ull);           //     32768
  _Float16* wp1 = (_Float16*)(ws + 404819968ull);           //     32768
  _Float16* wp2 = (_Float16*)(ws + 404852736ull);           //     65536

  float* new_xyz = out;                 // [16][1024][3]
  float* tf      = out + 16*1024*3;     // [16][256][2]

  fps_kernel   <<<NB, 256, 0, stream>>>(xyz, new_xyz);
  knn_kernel   <<<NB*NPOINT, 32, 0, stream>>>(xyz, new_xyz, idx);
  gather_kernel<<<NB*(MTOT/32), 256, 0, stream>>>(points, idx, FA);

  wprep_kernel <<<64, 256, 0, stream>>>(W0, wp0);    // 128*128 elements
  wprep_kernel <<<64, 256, 0, stream>>>(W1, wp1);
  wprep_kernel <<<128, 256, 0, stream>>>(W2, wp2);   // 256*128 elements

  // layer 1: FA(128) -> FB(128), BN in place on FB
  zero_kernel  <<<2, 256, 0, stream>>>(stats, 512);
  mm_kernel    <<<NB*8*512/8, 256, 0, stream>>>(wp0, b0, FA, FB, 128);
  stats_kernel <<<4*128, 256, 0, stream>>>(FB, 4, stats);
  bnprep_kernel<<<1, 256, 0, stream>>>(stats, g0, bb0, coef, 128);
  bn_kernel    <<<16384, 256, 0, stream>>>(FB, coef, 4);

  // layer 2: FB(128) -> FA(128), BN in place on FA
  zero_kernel  <<<2, 256, 0, stream>>>(stats, 512);
  mm_kernel    <<<NB*8*512/8, 256, 0, stream>>>(wp1, b1, FB, FA, 128);
  stats_kernel <<<4*128, 256, 0, stream>>>(FA, 4, stats);
  bnprep_kernel<<<1, 256, 0, stream>>>(stats, g1, bb1, coef, 128);
  bn_kernel    <<<16384, 256, 0, stream>>>(FA, coef, 4);

  // layer 3: FA(128) -> FB(256), BN in place on FB
  zero_kernel  <<<2, 256, 0, stream>>>(stats, 512);
  mm_kernel    <<<NB*16*512/8, 256, 0, stream>>>(wp2, b2, FA, FB, 256);
  stats_kernel <<<8*128, 256, 0, stream>>>(FB, 8, stats);
  bnprep_kernel<<<1, 256, 0, stream>>>(stats, g2, bb2, coef, 256);
  bn_kernel    <<<32768, 256, 0, stream>>>(FB, coef, 8);

  // pool
  zero_kernel  <<<32, 256, 0, stream>>>(pstat, 8192);
  poolacc_kernel<<<NB*8*16, 256, 0, stream>>>(FB, pstat);
  poolfin_kernel<<<NB, 256, 0, stream>>>(pstat, tf);
}
